// MaxState_42958262894626
// MI455X (gfx1250) — matrix operand order repaired and checked by reading, not compiled, not run on hardware
//
#include <hip/hip_runtime.h>

// ---------------------------------------------------------------------------
// MaxState fused pipeline for gfx1250 (MI455X)
//   out0/1/2 = x @ W{0,1,2}^T   via bf16-split WMMA (hi/lo error compensation)
//   cm       = cummax_s((out0+out1)/8)
//   out      = (cm + out2) * cm + out1
// ---------------------------------------------------------------------------

typedef __attribute__((ext_vector_type(16))) __bf16 v16bf;
typedef __attribute__((ext_vector_type(8)))  float  v8f;
typedef __attribute__((__vector_size__(16))) int    i32x4;

#define HIDDEN 1024
#define SEQ    4096
#define BATCH  4
#define MTOT   (BATCH * SEQ)      // 16384 rows
#define KDIM   HIDDEN
#define NDIM   HIDDEN

#define MT 128                     // block tile M
#define NT 128                     // block tile N
#define KC 32                      // K chunk (one bf16 WMMA K)

#define CHUNK   256                // scan chunk length
#define NCHUNK  (SEQ / CHUNK)      // 16
#define NCOL    (BATCH * HIDDEN)   // 4096 independent scan columns

// ---------------------------------------------------------------------------
// CDNA5 async global->LDS copy (ASYNCcnt) with register-staged fallback.
// Builtin signature (from hipcc diagnostic): first param is
//   __attribute__((__vector_size__(16))) int * in the global address space.
// ---------------------------------------------------------------------------
#if defined(__HIP_DEVICE_COMPILE__) && __has_builtin(__builtin_amdgcn_global_load_async_to_lds_b128)
#define HAVE_ASYNC_LDS 1
#else
#define HAVE_ASYNC_LDS 0
#endif

static __device__ __forceinline__ void async_copy_b128(const void* g, void* l) {
#if HAVE_ASYNC_LDS
    __builtin_amdgcn_global_load_async_to_lds_b128(
        (__attribute__((address_space(1))) i32x4*)g,
        (__attribute__((address_space(3))) i32x4*)l,
        0, 0);
#else
    *(uint4*)l = *(const uint4*)g;
#endif
}

static __device__ __forceinline__ void async_wait_all() {
#if HAVE_ASYNC_LDS
#if __has_builtin(__builtin_amdgcn_s_wait_asynccnt)
    __builtin_amdgcn_s_wait_asynccnt(0);
#else
    asm volatile("s_wait_asynccnt 0" ::: "memory");
#endif
#endif
}

// ---------------------------------------------------------------------------
// fp32 -> bf16 hi/lo split (round-to-nearest-even)
// ---------------------------------------------------------------------------
static __device__ __forceinline__ unsigned short f32_to_bf16_rn(float f) {
    unsigned int u = __float_as_uint(f);
    u += 0x7fffu + ((u >> 16) & 1u);
    return (unsigned short)(u >> 16);
}

__global__ void split_bf16_kernel(const float* __restrict__ src,
                                  unsigned short* __restrict__ hi,
                                  unsigned short* __restrict__ lo,
                                  int n) {
    int i = blockIdx.x * blockDim.x + threadIdx.x;
    if (i >= n) return;
    float v = src[i];
    unsigned short h = f32_to_bf16_rn(v);
    float hf = __uint_as_float(((unsigned int)h) << 16);
    hi[i] = h;
    lo[i] = f32_to_bf16_rn(v - hf);
}

// ---------------------------------------------------------------------------
// GEMM: P[w] = x @ W[w]^T with bf16x3 split accumulation in f32 WMMA.
// Block = 256 threads = 8 waves arranged 4(M) x 2(N); each wave owns a
// 32x64 output region = 2x4 subtiles of 16x16.
// Double-buffered LDS; async DMA for chunk k+1 overlaps WMMAs of chunk k.
// ---------------------------------------------------------------------------
__global__ void __launch_bounds__(256)
gemm_bf16x3_kernel(const unsigned short* __restrict__ xhi,
                   const unsigned short* __restrict__ xlo,
                   const unsigned short* __restrict__ whi,   // [3][N][K]
                   const unsigned short* __restrict__ wlo,   // [3][N][K]
                   float* __restrict__ P)                    // [3][M][N]
{
    __shared__ unsigned short lA[2][2][MT * KC];   // [buf][hi/lo]  2*16 KB
    __shared__ unsigned short lB[2][2][NT * KC];   // [buf][hi/lo]  2*16 KB

    const int mt   = blockIdx.x;      // 0..127
    const int nt   = blockIdx.y;      // 0..7
    const int w    = blockIdx.z;      // 0..2
    const int tid  = threadIdx.x;
    const int lane = tid & 31;
    const int wave = tid >> 5;
    const int waveM = wave & 3;       // 0..3  -> 32 rows each
    const int waveN = wave >> 2;      // 0..1  -> 64 cols each

    const unsigned short* wh = whi + (size_t)w * NDIM * KDIM;
    const unsigned short* wl = wlo + (size_t)w * NDIM * KDIM;
    float* Pw = P + (size_t)w * MTOT * NDIM;

    const int gm0 = mt * MT;
    const int gn0 = nt * NT;

    v8f acc[2][4];
    #pragma unroll
    for (int i = 0; i < 2; ++i)
        #pragma unroll
        for (int j = 0; j < 4; ++j)
            acc[i][j] = (v8f){};

    const int laneM = lane & 15;
    const int hiHalf = lane >> 4;     // 0 for lanes 0-15, 1 for lanes 16-31

    // Stage one K-chunk (4 tiles of 128x32 bf16) into LDS buffer `buf`.
    // 4096 ushorts per tile = 512 x 16B chunks; 2 chunks per thread.
    auto stage = [&](int buf, int kc2) {
        #pragma unroll
        for (int c = 0; c < 2; ++c) {
            int idx = tid + c * 256;          // 0..511
            int r   = idx >> 2;               // tile row 0..127
            int cc  = idx & 3;                // 8-ushort chunk in row
            size_t gx = (size_t)(gm0 + r) * KDIM + kc2 + cc * 8;
            size_t gw = (size_t)(gn0 + r) * KDIM + kc2 + cc * 8;
            int    lo = r * KC + cc * 8;
            async_copy_b128(xhi + gx, &lA[buf][0][lo]);
            async_copy_b128(xlo + gx, &lA[buf][1][lo]);
            async_copy_b128(wh  + gw, &lB[buf][0][lo]);
            async_copy_b128(wl  + gw, &lB[buf][1][lo]);
        }
    };

    // Prologue: fill buffer 0.
    stage(0, 0);
    async_wait_all();
    __syncthreads();

    for (int kc = 0; kc < KDIM; kc += KC) {
        const int cur = (kc / KC) & 1;

        // Kick off DMA for the next chunk into the other buffer.
        if (kc + KC < KDIM) stage(cur ^ 1, kc + KC);

        // ---- build fragments, same contiguous k order in A and B ----------
        union Frag { v16bf v; unsigned int u[8]; };

        // A: 16x32, lane = M row; lanes 0-15 hold K 0..15,
        //           lanes 16-31 hold K 16..31.
        Frag ahi[2], alo[2];
        #pragma unroll
        for (int ms = 0; ms < 2; ++ms) {
            int row = waveM * 32 + ms * 16 + laneM;
            const unsigned int* rAh = (const unsigned int*)(&lA[cur][0][row * KC]);
            const unsigned int* rAl = (const unsigned int*)(&lA[cur][1][row * KC]);
            #pragma unroll
            for (int j = 0; j < 8; ++j) {
                int k0 = j * 2 + (hiHalf ? 16 : 0);
                ahi[ms].u[j] = rAh[k0 >> 1];
                alo[ms].u[j] = rAl[k0 >> 1];
            }
        }

        // B: 32x16, lane%16 = N col; lanes 0-15 hold K 0..15,
        //           lanes 16-31 hold K 16..31, two K per VGPR.
        #pragma unroll
        for (int ns = 0; ns < 4; ++ns) {
            Frag bhi, blo;
            int nrow = waveN * 64 + ns * 16 + laneM;
            int kb   = hiHalf ? 16 : 0;
            const unsigned int* rBh = (const unsigned int*)(&lB[cur][0][nrow * KC + kb]);
            const unsigned int* rBl = (const unsigned int*)(&lB[cur][1][nrow * KC + kb]);
            #pragma unroll
            for (int j = 0; j < 8; ++j) { bhi.u[j] = rBh[j]; blo.u[j] = rBl[j]; }

            #pragma unroll
            for (int ms = 0; ms < 2; ++ms) {
                acc[ms][ns] = __builtin_amdgcn_wmma_f32_16x16x32_bf16(
                    false, ahi[ms].v, false, bhi.v, (short)0, acc[ms][ns], false, false);
                acc[ms][ns] = __builtin_amdgcn_wmma_f32_16x16x32_bf16(
                    false, ahi[ms].v, false, blo.v, (short)0, acc[ms][ns], false, false);
                acc[ms][ns] = __builtin_amdgcn_wmma_f32_16x16x32_bf16(
                    false, alo[ms].v, false, bhi.v, (short)0, acc[ms][ns], false, false);
            }
        }

        // DMA for next chunk complete + everyone done reading current chunk.
        async_wait_all();
        __syncthreads();
    }

    // ---- epilogue: C layout VGPR v -> M = v + 8*hiHalf, col = lane%16 ----
    #pragma unroll
    for (int ms = 0; ms < 2; ++ms)
        #pragma unroll
        for (int ns = 0; ns < 4; ++ns)
            #pragma unroll
            for (int v = 0; v < 8; ++v) {
                int row = gm0 + waveM * 32 + ms * 16 + v + (hiHalf << 3);
                int col = gn0 + waveN * 64 + ns * 16 + laneM;
                Pw[(size_t)row * NDIM + col] = acc[ms][ns][v];
            }
}

// ---------------------------------------------------------------------------
// Decoupled cummax scan over the sequence axis, channel-coalesced.
// t = ((b * NCHUNK + j) * HIDDEN + n)
// ---------------------------------------------------------------------------
__global__ void scan_pass1(const float* __restrict__ P,
                           float* __restrict__ partials) {
    int t = blockIdx.x * blockDim.x + threadIdx.x;
    if (t >= NCOL * NCHUNK) return;
    int n = t & (HIDDEN - 1);
    int j = (t >> 10) & (NCHUNK - 1);
    int b = t >> 14;
    const float* P0 = P;
    const float* P1 = P + (size_t)MTOT * NDIM;
    size_t base = ((size_t)b * SEQ + (size_t)j * CHUNK) * HIDDEN + n;
    float m = -3.402823466e+38f;
    for (int s = 0; s < CHUNK; ++s) {
        float v = (P0[base] + P1[base]) * 0.125f;   // 1/sqrt(64)
        m = fmaxf(m, v);
        base += HIDDEN;
    }
    partials[t] = m;
}

__global__ void scan_pass2(const float* __restrict__ partials,
                           float* __restrict__ carries) {
    int t = blockIdx.x * blockDim.x + threadIdx.x;  // 0..NCOL-1
    if (t >= NCOL) return;
    int n = t & (HIDDEN - 1);
    int b = t >> 10;
    float carry = -3.402823466e+38f;
    for (int j = 0; j < NCHUNK; ++j) {
        int idx = ((b * NCHUNK + j) << 10) | n;
        carries[idx] = carry;                        // exclusive prefix max
        carry = fmaxf(carry, partials[idx]);
    }
}

__global__ void scan_pass3(const float* __restrict__ P,
                           const float* __restrict__ carries,
                           float* __restrict__ out) {
    int t = blockIdx.x * blockDim.x + threadIdx.x;
    if (t >= NCOL * NCHUNK) return;
    int n = t & (HIDDEN - 1);
    int j = (t >> 10) & (NCHUNK - 1);
    int b = t >> 14;
    const float* P0 = P;
    const float* P1 = P + (size_t)MTOT * NDIM;
    const float* P2 = P + 2 * (size_t)MTOT * NDIM;
    size_t base = ((size_t)b * SEQ + (size_t)j * CHUNK) * HIDDEN + n;
    float m = carries[t];
    for (int s = 0; s < CHUNK; ++s) {
        float v = (P0[base] + P1[base]) * 0.125f;
        m = fmaxf(m, v);
        float o1 = P1[base];
        out[base] = (m + P2[base]) * m + o1;
        base += HIDDEN;
    }
}

// ---------------------------------------------------------------------------
extern "C" void kernel_launch(void* const* d_in, const int* in_sizes, int n_in,
                              void* d_out, int out_size, void* d_ws, size_t ws_size,
                              hipStream_t stream) {
    const float* x  = (const float*)d_in[0];
    const float* W0 = (const float*)d_in[1];
    const float* W1 = (const float*)d_in[2];
    const float* W2 = (const float*)d_in[3];

    char* ws = (char*)d_ws;
    size_t off = 0;
    unsigned short* xhi = (unsigned short*)(ws + off); off += (size_t)MTOT * KDIM * 2;
    unsigned short* xlo = (unsigned short*)(ws + off); off += (size_t)MTOT * KDIM * 2;
    unsigned short* whi = (unsigned short*)(ws + off); off += (size_t)3 * NDIM * KDIM * 2;
    unsigned short* wlo = (unsigned short*)(ws + off); off += (size_t)3 * NDIM * KDIM * 2;
    float* P        = (float*)(ws + off); off += (size_t)3 * MTOT * NDIM * 4;
    float* partials = (float*)(ws + off); off += (size_t)NCOL * NCHUNK * 4;
    float* carries  = (float*)(ws + off); off += (size_t)NCOL * NCHUNK * 4;

    // 1) fp32 -> bf16 hi/lo splits
    int nx = MTOT * KDIM;
    int nw = NDIM * KDIM;
    split_bf16_kernel<<<(nx + 255) / 256, 256, 0, stream>>>(x,  xhi, xlo, nx);
    split_bf16_kernel<<<(nw + 255) / 256, 256, 0, stream>>>(W0, whi,          wlo,          nw);
    split_bf16_kernel<<<(nw + 255) / 256, 256, 0, stream>>>(W1, whi + nw,     wlo + nw,     nw);
    split_bf16_kernel<<<(nw + 255) / 256, 256, 0, stream>>>(W2, whi + 2 * nw, wlo + 2 * nw, nw);

    // 2) three GEMMs via grid.z; weights + x(bf16) fit in 192 MB L2
    dim3 gg(MTOT / MT, NDIM / NT, 3);
    gemm_bf16x3_kernel<<<gg, 256, 0, stream>>>(xhi, xlo, whi, wlo, P);

    // 3) decoupled cummax + fused epilogue
    int np = NCOL * NCHUNK;
    scan_pass1<<<np / 256, 256, 0, stream>>>(P, partials);
    scan_pass2<<<NCOL / 256, 256, 0, stream>>>(partials, carries);
    scan_pass3<<<np / 256, 256, 0, stream>>>(P, carries, (float*)d_out);
}
